// ExpKernel_2886218023609
// MI455X (gfx1250) — matrix operand order repaired and checked by reading, not compiled, not run on hardware
//
#include <hip/hip_runtime.h>
#include <hip/hip_bf16.h>

// ---------------------------------------------------------------------------
// A = exp(-sqrt(1e-6 + ||y_i - y_j||^2)/10), Y = MLP(x), N=8192
// f16 WMMA everywhere; norms+sqrt+exp fused into the Gram epilogue.
// Gram waves own 32x128 strips so every B-fragment feeds 2 WMMAs.
// Tile staging uses GLOBAL_LOAD_ASYNC_TO_LDS_B128 (ASYNCcnt path) when the
// builtin is available.
// ---------------------------------------------------------------------------

typedef __attribute__((ext_vector_type(16))) _Float16 v16h;
typedef __attribute__((ext_vector_type(8)))  _Float16 v8h;
typedef __attribute__((ext_vector_type(8)))  float    v8f;
typedef __attribute__((ext_vector_type(4)))  int      v4i;

union V16U { v16h v; v8h h[2]; };

__device__ __forceinline__ v16h make_v16(v8h lo, v8h hi) {
    V16U u; u.h[0] = lo; u.h[1] = hi; return u.v;
}

// A-fragment (16x32 f16, M x K): lane L holds row M = L%16;
// VGPR0..3 = K kb..kb+7, VGPR4..7 = K kb+16..kb+23 with kb=(L/16)*8.
__device__ __forceinline__ v16h load_a_frag(const _Float16* base, int stride_h, int lane) {
    const _Float16* p = base + (lane & 15) * stride_h + ((lane >> 4) * 8);
    v8h lo = *(const v8h*)(p);
    v8h hi = *(const v8h*)(p + 16);
    return make_v16(lo, hi);
}

// B-fragment (32x16 f16, K x N): lane L holds column n = L%16;
// VGPR0..3 = K kb..kb+7, VGPR4..7 = K kb+16..kb+23 with kb=(L/16)*8.
__device__ __forceinline__ v16h load_b_frag(const _Float16* base, int stride_h, int lane) {
    const _Float16* p = base + (lane & 15) * stride_h + ((lane >> 4) * 8);
    v8h lo = *(const v8h*)(p);
    v8h hi = *(const v8h*)(p + 16);
    return make_v16(lo, hi);
}

// ---- CDNA5 async global->LDS staging (ASYNCcnt-tracked, no VGPR data) ------
#if defined(__HIP_DEVICE_COMPILE__) && __has_builtin(__builtin_amdgcn_global_load_async_to_lds_b128)
#define HAVE_ASYNC_LDS 1
__device__ __forceinline__ void async_copy16(const void* gsrc, void* ldst) {
    __builtin_amdgcn_global_load_async_to_lds_b128(
        (__attribute__((address_space(1))) v4i*)gsrc,
        (__attribute__((address_space(3))) v4i*)ldst, 0, 0);
}
__device__ __forceinline__ void async_wait0() {
#if __has_builtin(__builtin_amdgcn_s_wait_asynccnt)
    __builtin_amdgcn_s_wait_asynccnt(0);
#else
    asm volatile("s_wait_asynccnt 0x0" ::: "memory");
#endif
}
#else
#define HAVE_ASYNC_LDS 0
#endif

// ---------------------------------------------------------------------------
// Kernel 1: convert W (KxN f32 row-major) -> Wt (NxK f16 row-major).
// ---------------------------------------------------------------------------
__global__ void wcvt_kernel(const float* __restrict__ W, _Float16* __restrict__ Wt,
                            int K, int Nn) {
    int idx = blockIdx.x * blockDim.x + threadIdx.x;
    int total = K * Nn;
    if (idx < total) {
        int n = idx / K;
        int k = idx - n * K;
        Wt[idx] = (_Float16)W[k * Nn + n];
    }
}

// ---------------------------------------------------------------------------
// Kernel 2: fused 3-layer MLP, one wave per 16-row strip.
// ---------------------------------------------------------------------------
#define SX 520   // LDS row stride (halves) for 512-wide buffers (bank-pad +8)

template <int K, int NT>
__device__ __forceinline__ void mlp_layer(const _Float16* __restrict__ inBuf, int inStride,
                                          const _Float16* __restrict__ Wt,
                                          const float* __restrict__ bias,
                                          _Float16* __restrict__ outBuf, int outStride,
                                          bool relu, int lane) {
    const int n16 = lane & 15;
    const int mhi = (lane >> 4) * 8;
    for (int nt = 0; nt < NT; ++nt) {
        float bv = bias[nt * 16 + n16];
        v8f c;
        #pragma unroll
        for (int i = 0; i < 8; ++i) c[i] = bv;
        #pragma unroll 4
        for (int kt = 0; kt < K / 32; ++kt) {
            v16h a = load_a_frag(inBuf + kt * 32, inStride, lane);
            v16h b = load_b_frag(Wt + (nt * 16) * K + kt * 32, K, lane);
            c = __builtin_amdgcn_wmma_f32_16x16x32_f16(false, a, false, b,
                                                       (short)0, c, false, false);
        }
        #pragma unroll
        for (int r = 0; r < 8; ++r) {
            float v = c[r];
            if (relu) v = fmaxf(v, 0.0f);
            int M = r + mhi;
            outBuf[M * outStride + nt * 16 + n16] = (_Float16)v;
        }
    }
}

__global__ __launch_bounds__(32) void mlp_kernel(const float* __restrict__ x,
                                                 const _Float16* __restrict__ W1t,
                                                 const float* __restrict__ b1,
                                                 const _Float16* __restrict__ W2t,
                                                 const float* __restrict__ b2,
                                                 const _Float16* __restrict__ W3t,
                                                 const float* __restrict__ b3,
                                                 _Float16* __restrict__ Yh,
                                                 float* __restrict__ norms) {
    __shared__ __align__(16) _Float16 bufA[16 * SX];
    __shared__ __align__(16) _Float16 bufB[16 * SX];
    const int lane = threadIdx.x;
    const int row0 = blockIdx.x * 16;

    for (int r = 0; r < 16; ++r) {
        const float4* src = (const float4*)(x + (size_t)(row0 + r) * 512);
        #pragma unroll
        for (int cidx = 0; cidx < 4; ++cidx) {
            float4 v = src[cidx * 32 + lane];
            _Float16* d = &bufA[r * SX + cidx * 128 + lane * 4];
            d[0] = (_Float16)v.x; d[1] = (_Float16)v.y;
            d[2] = (_Float16)v.z; d[3] = (_Float16)v.w;
        }
    }
    __syncthreads();

    mlp_layer<512, 32>(bufA, SX, W1t, b1, bufB, SX, true,  lane);
    __syncthreads();
    mlp_layer<512, 16>(bufB, SX, W2t, b2, bufA, SX, true,  lane);
    __syncthreads();
    mlp_layer<256,  8>(bufA, SX, W3t, b3, bufB, SX, false, lane);
    __syncthreads();

    for (int idx = lane; idx < 16 * 8; idx += 32) {
        int r = idx >> 3, seg = idx & 7;
        *(v8h*)(Yh + (size_t)(row0 + r) * 128 + seg * 8) =
            *(const v8h*)(&bufB[r * SX + seg * 8]);
    }
    if (lane < 16) {
        float s = 0.0f;
        #pragma unroll 8
        for (int kk = 0; kk < 128; ++kk) {
            float y = (float)bufB[lane * SX + kk];
            s = fmaf(y, y, s);
        }
        norms[row0 + lane] = s;
    }
}

// ---------------------------------------------------------------------------
// Kernel 3: 256x128 Gram tile per 8-wave WG; each wave owns a 32x128 strip
// (two M-tiles share each B-fragment).  Fused distance/exp epilogue.
// ---------------------------------------------------------------------------
#define SJ 136   // LDS row stride (halves) for 128-wide Y tiles (bank-pad +8)

__global__ __launch_bounds__(256) void gram_kernel(const _Float16* __restrict__ Yh,
                                                   const float* __restrict__ norms,
                                                   float* __restrict__ out) {
    extern __shared__ char smem[];
    _Float16* sYi = (_Float16*)smem;                 // 256 x SJ
    _Float16* sYj = sYi + 256 * SJ;                  // 128 x SJ
    float*    sNi = (float*)(sYj + 128 * SJ);        // 256
    float*    sNj = sNi + 256;                       // 128

    const int tid  = threadIdx.x;
    const int lane = tid & 31;
    const int w    = tid >> 5;               // wave id, 0..7
    const int j0   = blockIdx.x * 128;
    const int i0   = blockIdx.y * 256;

    // Stage Yi (256x128) and Yj (128x128) f16 tiles: 384 rows x 8 x 16B chunks.
    #pragma unroll
    for (int it = 0; it < 12; ++it) {
        int idx = tid + it * 256;
        int r = idx >> 3, seg = idx & 7;
        const _Float16* gsrc;
        _Float16* ldst;
        if (r < 256) {
            gsrc = Yh + (size_t)(i0 + r) * 128 + seg * 8;
            ldst = &sYi[r * SJ + seg * 8];
        } else {
            int rr = r - 256;
            gsrc = Yh + (size_t)(j0 + rr) * 128 + seg * 8;
            ldst = &sYj[rr * SJ + seg * 8];
        }
#if HAVE_ASYNC_LDS
        async_copy16(gsrc, ldst);
#else
        *(v8h*)ldst = *(const v8h*)gsrc;
#endif
    }
    sNi[tid] = norms[i0 + tid];
    if (tid < 128) sNj[tid] = norms[j0 + tid];
#if HAVE_ASYNC_LDS
    async_wait0();
#endif
    __syncthreads();

    v8f c[2][8];
    #pragma unroll
    for (int mt = 0; mt < 2; ++mt)
        #pragma unroll
        for (int nt = 0; nt < 8; ++nt)
            #pragma unroll
            for (int i = 0; i < 8; ++i) c[mt][nt][i] = 0.0f;

    const int nlane = lane & 15;
    #pragma unroll
    for (int kt = 0; kt < 4; ++kt) {
        v16h a0 = load_a_frag(&sYi[(w * 32 +  0) * SJ + kt * 32], SJ, lane);
        v16h a1 = load_a_frag(&sYi[(w * 32 + 16) * SJ + kt * 32], SJ, lane);
        #pragma unroll
        for (int nt = 0; nt < 8; ++nt) {
            v16h b = load_b_frag(&sYj[(nt * 16) * SJ + kt * 32], SJ, lane);
            c[0][nt] = __builtin_amdgcn_wmma_f32_16x16x32_f16(false, a0, false, b,
                                                              (short)0, c[0][nt], false, false);
            c[1][nt] = __builtin_amdgcn_wmma_f32_16x16x32_f16(false, a1, false, b,
                                                              (short)0, c[1][nt], false, false);
        }
    }

    // Epilogue: dist2 = n_i + n_j - 2*g ; A = exp(-sqrt(1e-6+dist2)/10)
    const int mhi = (lane >> 4) * 8;
    #pragma unroll
    for (int mt = 0; mt < 2; ++mt) {
        #pragma unroll
        for (int nt = 0; nt < 8; ++nt) {
            float nj = sNj[nt * 16 + nlane];
            #pragma unroll
            for (int r = 0; r < 8; ++r) {
                int M = w * 32 + mt * 16 + r + mhi;
                float d2 = sNi[M] + nj - 2.0f * c[mt][nt][r];
                d2 = fmaxf(d2, 0.0f);
                float dist = sqrtf(1e-6f + d2);
                out[(size_t)(i0 + M) * 8192 + (j0 + nt * 16 + nlane)] = __expf(dist * -0.1f);
            }
        }
    }
}

// ---------------------------------------------------------------------------
// Host launcher
// ---------------------------------------------------------------------------
extern "C" void kernel_launch(void* const* d_in, const int* in_sizes, int n_in,
                              void* d_out, int out_size, void* d_ws, size_t ws_size,
                              hipStream_t stream) {
    const float* x  = (const float*)d_in[0];
    const float* W1 = (const float*)d_in[1];
    const float* b1 = (const float*)d_in[2];
    const float* W2 = (const float*)d_in[3];
    const float* b2 = (const float*)d_in[4];
    const float* W3 = (const float*)d_in[5];
    const float* b3 = (const float*)d_in[6];
    float* out = (float*)d_out;

    char* ws = (char*)d_ws;
    _Float16* Yh    = (_Float16*)ws;                                   // 2 MB
    float*    norms = (float*)(ws + 2u * 1024u * 1024u);               // 32 KB
    _Float16* W1t   = (_Float16*)(ws + 2u * 1024u * 1024u + 64u * 1024u);
    _Float16* W2t   = W1t + 512 * 512;
    _Float16* W3t   = W2t + 256 * 512;

    wcvt_kernel<<<(512 * 512 + 255) / 256, 256, 0, stream>>>(W1, W1t, 512, 512);
    wcvt_kernel<<<(512 * 256 + 255) / 256, 256, 0, stream>>>(W2, W2t, 512, 256);
    wcvt_kernel<<<(256 * 128 + 255) / 256, 256, 0, stream>>>(W3, W3t, 256, 128);

    mlp_kernel<<<8192 / 16, 32, 0, stream>>>(x, W1t, b1, W2t, b2, W3t, b3, Yh, norms);

    dim3 grid(8192 / 128, 8192 / 256);
    size_t smem = (size_t)(256 + 128) * SJ * sizeof(_Float16)
                + (size_t)(256 + 128) * sizeof(float);
    gram_kernel<<<grid, 256, smem, stream>>>(Yh, norms, out);
}
